// FusionKANLayer_19035295056670
// MI455X (gfx1250) — matrix operand order repaired and checked by reading, not compiled, not run on hardware
//
#include <hip/hip_runtime.h>
#include <hip/hip_bf16.h>

// ---------------------------------------------------------------------------
// Fused KAN layer for MI455X (gfx1250, wave32, WMMA).
//
// y = [ SiLU(x)@Wb^T*sb + bias*sb ] + [ einsum(bases, Ws)*ss ]  -> LN -> PReLU
//
// Reformulated as ONE bf16 GEMM:  A[4096, 9216] @ W[9216, 1024]
//   K-block 0       : A = SiLU(x[b,i]),        W = Wb[o,i]*scale_base[o]
//   K-block c=1..8  : A = basis_{c-1}(x[b,i]), W = Ws[o,i,c-1]*scale_spline[o]
// 77 GFLOP vs ~95MB staged data (L2-resident; L2=192MB) -> compute bound ->
// drive it through v_wmma_f32_16x16x32_bf16 (f32 accumulate).
//
// Tile staging via GLOBAL_LOAD_ASYNC_TO_LDS_B128 (ASYNCcnt) — builtin takes
// (int4 AS(1)*, int4 AS(3)*, imm offset, imm cpol) per round-2 diagnostics.
//
// Workspace: A-frags 75.5MB + B-frags 18.9MB = 94.4MB (bf16, WMMA lane layout).
// ---------------------------------------------------------------------------

typedef __attribute__((ext_vector_type(16))) __bf16 v16bf;
typedef __attribute__((ext_vector_type(8)))  float  v8f;
typedef int v4i_raw __attribute__((__vector_size__(4 * sizeof(int))));

#define B_ROWS 4096
#define IN_F   1024
#define OUT_F  1024
#define KDIM   (IN_F * 9)    // 9216
#define KT     (KDIM / 32)   // 288 K-tiles of 32

#if __has_builtin(__builtin_amdgcn_global_load_async_to_lds_b128) && \
    __has_builtin(__builtin_amdgcn_s_wait_asynccnt)
#define USE_ASYNC_LDS 1
#else
#define USE_ASYNC_LDS 0
#endif

#if USE_ASYNC_LDS
#define ASYNC_CP_B128(gp, lp)                                         \
  __builtin_amdgcn_global_load_async_to_lds_b128(                     \
      (__attribute__((address_space(1))) v4i_raw*)(gp),               \
      (__attribute__((address_space(3))) v4i_raw*)(lp), 0, 0)
#endif

__device__ __forceinline__ unsigned short f2bf(float f) {
  unsigned u = __float_as_uint(f);
  u += 0x7fffu + ((u >> 16) & 1u);      // round-to-nearest-even
  return (unsigned short)(u >> 16);
}

// SiLU + all 8 cubic B-spline bases (Cox-de Boor, uniform knots h=0.4)
__device__ __forceinline__ void kan_feats(float x, float* f) {
  const float h = 0.4f;
  float kn[12];
#pragma unroll
  for (int t = 0; t < 12; ++t) kn[t] = -1.0f + h * (float)(t - 3);
  float b[11];
#pragma unroll
  for (int j = 0; j < 11; ++j) b[j] = (x >= kn[j] && x < kn[j + 1]) ? 1.0f : 0.0f;
#pragma unroll
  for (int p = 1; p <= 3; ++p) {
    float inv = 1.0f / (h * (float)p);
#pragma unroll
    for (int j = 0; j < 11 - p; ++j)
      b[j] = (x - kn[j]) * inv * b[j] + (kn[j + p + 1] - x) * inv * b[j + 1];
  }
  f[0] = x / (1.0f + __expf(-x));        // SiLU
#pragma unroll
  for (int j = 0; j < 8; ++j) f[1 + j] = b[j];
}

// --- Kernel 1: merged/scaled weights -> WMMA B-fragment layout (bf16) ------
// B 32x16 bf16 layout (ISA 7.12.2): lane = 16*((k>>3)&1) + n, elem = 8*(k>>4) + (k&7).
__global__ __launch_bounds__(256) void kan_merge_w(
    const float* __restrict__ Wb, const float* __restrict__ Ws,
    const float* __restrict__ sb, const float* __restrict__ ss,
    unsigned short* __restrict__ Bf) {
  int gid = blockIdx.x * 256 + threadIdx.x;     // one thread per (o, i)
  int i = gid & (IN_F - 1);
  int o = gid >> 10;
  int nt = o >> 4, nl = o & 15;
  int kl = i & 31;
  int e    = (kl >> 4) * 8 + (kl & 7);
  int lane = ((kl >> 3) & 1) * 16 + nl;
  size_t row = ((size_t)nt * KT) * 32 + lane;
  {
    int kt = (i >> 5);                          // K-block 0: base weight
    Bf[(row + (size_t)kt * 32) * 16 + e] = f2bf(Wb[gid] * sb[o]);
  }
  float s = ss[o];
#pragma unroll
  for (int c = 0; c < 8; ++c) {                 // K-blocks 1..8: spline coeffs
    int kt = (c + 1) * 32 + (i >> 5);
    Bf[(row + (size_t)kt * 32) * 16 + e] = f2bf(Ws[(size_t)gid * 8 + c] * s);
  }
}

// --- Kernel 2: activation expansion -> WMMA A-fragment layout (bf16) -------
// A 16x32 bf16 layout (ISA 7.12.2): k = 16*(v>>2) + 8*(lane>>4) + 2*(v&3) + el
__global__ __launch_bounds__(256) void kan_expand(
    const float* __restrict__ x, unsigned short* __restrict__ Af) {
  int gid = blockIdx.x * 256 + threadIdx.x;     // one thread per (b, i)
  int i = gid & (IN_F - 1);
  int brow = gid >> 10;
  float f[9];
  kan_feats(x[gid], f);
  int mt = brow >> 4, ml = brow & 15;
  int kl = i & 31;
  int group = kl >> 4, rem = kl & 15;
  int half = rem >> 3, jj = (rem >> 1) & 3, el = rem & 1;
  int e    = 2 * (group * 4 + jj) + el;
  int lane = half * 16 + ml;
  size_t row = ((size_t)mt * KT) * 32 + lane;
#pragma unroll
  for (int c = 0; c < 9; ++c) {
    int kt = c * 32 + (i >> 5);
    Af[(row + (size_t)kt * 32) * 16 + e] = f2bf(f[c]);
  }
}

// --- Kernel 3: bf16 WMMA GEMM, 128x128 tile / WG, LDS double-buffered ------
__global__ __launch_bounds__(256) void kan_gemm(
    const unsigned short* __restrict__ Af, const unsigned short* __restrict__ Bf,
    const float* __restrict__ bias, const float* __restrict__ sb,
    float* __restrict__ out) {
  __shared__ uint4 sA[2][512];                  // 8 frags * 32 lanes * 32B = 8KB/buf
  __shared__ uint4 sB[2][512];
  const uint4* Ag = (const uint4*)Af;
  const uint4* Bg = (const uint4*)Bf;

  int bm = blockIdx.y, bn = blockIdx.x;
  int t = threadIdx.x, lane = t & 31, wave = t >> 5;
  int wr = wave >> 2, wc = wave & 3;            // 2x4 wave grid -> 64x32 per wave
  int lfrag = t >> 5;                           // loader handles fragment lfrag

  size_t abase = ((size_t)(bm * 8 + lfrag) * KT) * 32 + lane;  // frag-lane units
  size_t bbase = ((size_t)(bn * 8 + lfrag) * KT) * 32 + lane;

  auto loadTiles = [&](int kt, int buf) {
    size_t ai = (abase + (size_t)kt * 32) * 2;  // uint4 units (32B per frag-lane)
    size_t bi = (bbase + (size_t)kt * 32) * 2;
#if USE_ASYNC_LDS
    // gfx1250 async DMA: global -> LDS directly, tracked with ASYNCcnt
    ASYNC_CP_B128(Ag + ai + 0, &sA[buf][t * 2 + 0]);
    ASYNC_CP_B128(Ag + ai + 1, &sA[buf][t * 2 + 1]);
    ASYNC_CP_B128(Bg + bi + 0, &sB[buf][t * 2 + 0]);
    ASYNC_CP_B128(Bg + bi + 1, &sB[buf][t * 2 + 1]);
#else
    sA[buf][t * 2 + 0] = Ag[ai + 0];
    sA[buf][t * 2 + 1] = Ag[ai + 1];
    sB[buf][t * 2 + 0] = Bg[bi + 0];
    sB[buf][t * 2 + 1] = Bg[bi + 1];
#endif
  };

  union Frag { uint4 u[2]; v16bf v; };
  v8f acc[4][2];
#pragma unroll
  for (int a = 0; a < 4; ++a)
#pragma unroll
    for (int b = 0; b < 2; ++b)
      acc[a][b] = (v8f){0.f, 0.f, 0.f, 0.f, 0.f, 0.f, 0.f, 0.f};

  loadTiles(0, 0);
#if USE_ASYNC_LDS
  __builtin_amdgcn_s_wait_asynccnt(0);
#endif
  __syncthreads();

  for (int kt = 0; kt < KT; ++kt) {
    int buf = kt & 1;
    if (kt + 1 < KT) loadTiles(kt + 1, buf ^ 1);  // overlaps with WMMAs below
    if (kt + 2 < KT) {                            // gfx1250 global_prefetch_b8
      __builtin_prefetch(Af + (abase + (size_t)(kt + 2) * 32) * 16, 0, 1);
      __builtin_prefetch(Bf + (bbase + (size_t)(kt + 2) * 32) * 16, 0, 1);
    }
    Frag fa[4], fb[2];
#pragma unroll
    for (int tm = 0; tm < 4; ++tm) {
      fa[tm].u[0] = sA[buf][(wr * 4 + tm) * 64 + lane * 2 + 0];
      fa[tm].u[1] = sA[buf][(wr * 4 + tm) * 64 + lane * 2 + 1];
    }
#pragma unroll
    for (int tn = 0; tn < 2; ++tn) {
      fb[tn].u[0] = sB[buf][(wc * 2 + tn) * 64 + lane * 2 + 0];
      fb[tn].u[1] = sB[buf][(wc * 2 + tn) * 64 + lane * 2 + 1];
    }
#pragma unroll
    for (int tm = 0; tm < 4; ++tm)
#pragma unroll
      for (int tn = 0; tn < 2; ++tn)
        acc[tm][tn] = __builtin_amdgcn_wmma_f32_16x16x32_bf16(
            false, fa[tm].v, false, fb[tn].v, (short)0, acc[tm][tn],
            false, false);
#if USE_ASYNC_LDS
    __builtin_amdgcn_s_wait_asynccnt(0);          // next-buf copies landed
#endif
    __syncthreads();
  }

  // epilogue: + bias*scale_base ; C/D layout: M = r + 8*(lane>>4), N = lane&15
  int col0 = bn * 128 + wc * 32;
  int row0 = bm * 128 + wr * 64;
  int hh = lane >> 4, nl = lane & 15;
#pragma unroll
  for (int tn = 0; tn < 2; ++tn) {
    int col = col0 + tn * 16 + nl;
    float cb = bias[col] * sb[col];
#pragma unroll
    for (int tm = 0; tm < 4; ++tm)
#pragma unroll
      for (int r = 0; r < 8; ++r) {
        int row = row0 + tm * 16 + hh * 8 + r;
        out[(size_t)row * OUT_F + col] = acc[tm][tn][r] + cb;
      }
  }
}

// --- Kernel 4: row LayerNorm + PReLU, in-place on d_out --------------------
__global__ __launch_bounds__(256) void kan_ln_prelu(
    float* __restrict__ y, const float* __restrict__ gamma,
    const float* __restrict__ beta, const float* __restrict__ pw) {
  __shared__ float rs[8], rq[8];
  int row = blockIdx.x, t = threadIdx.x;
  int lane = t & 31, wave = t >> 5;
  float4 v = ((const float4*)y)[(size_t)row * 256 + t];
  float s = v.x + v.y + v.z + v.w;
  float q = v.x * v.x + v.y * v.y + v.z * v.z + v.w * v.w;
#pragma unroll
  for (int off = 16; off >= 1; off >>= 1) {     // wave32 butterfly
    s += __shfl_xor(s, off, 32);
    q += __shfl_xor(q, off, 32);
  }
  if (lane == 0) { rs[wave] = s; rq[wave] = q; }
  __syncthreads();
  float ts = 0.f, tq = 0.f;
#pragma unroll
  for (int w = 0; w < 8; ++w) { ts += rs[w]; tq += rq[w]; }
  float mu   = ts * (1.0f / 1024.0f);
  float var  = tq * (1.0f / 1024.0f) - mu * mu;
  float rstd = rsqrtf(var + 1e-5f);
  float p = pw[0];
  float4 g  = ((const float4*)gamma)[t];
  float4 be = ((const float4*)beta)[t];
  float4 o;
  o.x = (v.x - mu) * rstd * g.x + be.x;
  o.y = (v.y - mu) * rstd * g.y + be.y;
  o.z = (v.z - mu) * rstd * g.z + be.z;
  o.w = (v.w - mu) * rstd * g.w + be.w;
  o.x = (o.x >= 0.f) ? o.x : p * o.x;
  o.y = (o.y >= 0.f) ? o.y : p * o.y;
  o.z = (o.z >= 0.f) ? o.z : p * o.z;
  o.w = (o.w >= 0.f) ? o.w : p * o.w;
  ((float4*)y)[(size_t)row * 256 + t] = o;
}

extern "C" void kernel_launch(void* const* d_in, const int* in_sizes, int n_in,
                              void* d_out, int out_size, void* d_ws, size_t ws_size,
                              hipStream_t stream) {
  (void)in_sizes; (void)n_in; (void)out_size; (void)ws_size;
  const float* x     = (const float*)d_in[0];
  const float* Ws    = (const float*)d_in[1];   // [OUT, IN, 8]
  const float* Wb    = (const float*)d_in[2];   // [OUT, IN]
  const float* bias  = (const float*)d_in[3];
  const float* sb    = (const float*)d_in[4];
  const float* ss    = (const float*)d_in[5];
  const float* gamma = (const float*)d_in[6];
  const float* beta  = (const float*)d_in[7];
  const float* pw    = (const float*)d_in[8];
  float* out = (float*)d_out;

  // workspace: A-frags [4096*9216] bf16, then B-frags [1024*9216] bf16 (94.4MB)
  unsigned short* Af = (unsigned short*)d_ws;
  unsigned short* Bf = Af + (size_t)B_ROWS * KDIM;

  kan_merge_w<<<(OUT_F * IN_F) / 256, 256, 0, stream>>>(Wb, Ws, sb, ss, Bf);
  kan_expand <<<(B_ROWS * IN_F) / 256, 256, 0, stream>>>(x, Af);
  dim3 grid(OUT_F / 128, B_ROWS / 128);         // 8 x 32 workgroups
  kan_gemm   <<<grid, 256, 0, stream>>>(Af, Bf, bias, sb, out);
  kan_ln_prelu<<<B_ROWS, 256, 0, stream>>>(out, gamma, beta, pw);
}
